// LLNPlusDiagAttention_12386685681796
// MI455X (gfx1250) — matrix operand order repaired and checked by reading, not compiled, not run on hardware
//
#include <hip/hip_runtime.h>
#include <math.h>

#define BATCH 4
#define NH 16
#define SEQ 4096
#define DIM 64
#define HEADS (BATCH * NH)   // 64
#define HSTR (SEQ * DIM)     // 262144 elements per head
#define CHUNKS 16
#define ROWP 72              // padded row stride (halves): 144 B, 16B-aligned
#define BLK_PER_WG 8         // fused kernel: token-blocks per workgroup
#define A_CONST 0.14855178144710912
#define B_CONST -0.35487039130661086
#define EPSF 1e-8f

typedef __attribute__((ext_vector_type(16))) _Float16 v16h;
typedef __attribute__((ext_vector_type(8)))  _Float16 v8h;
typedef __attribute__((ext_vector_type(4)))  _Float16 v4h;
typedef __attribute__((ext_vector_type(8)))  float    v8f;

__device__ __forceinline__ v8f wmma16x16x32(v16h a, v16h b, v8f c) {
  // D = A(16x32 f16) * B(32x16 f16) + C(16x16 f32)
  return __builtin_amdgcn_wmma_f32_16x16x32_f16(false, a, false, b, (short)0, c,
                                                false, false);
}

// Build v16h from two aligned 8-half (16B) LDS loads.
__device__ __forceinline__ v16h ld2x8(const _Float16* p0, const _Float16* p1) {
  v8h lo = *(const v8h*)p0;
  v8h hi = *(const v8h*)p1;
  return __builtin_shufflevector(lo, hi, 0, 1, 2, 3, 4, 5, 6, 7, 8, 9, 10, 11,
                                 12, 13, 14, 15);
}
// 16 contiguous halves (32B, aligned) -> v16h
__device__ __forceinline__ v16h ld16(const _Float16* p) {
  return ld2x8(p, p + 8);
}

// A-operand (16x32 f16) from an f32 row: per-lane runs of 8 at `base` and
// `base+16` (base = d0 + ((lane&16)?8:0)), scaled and converted to f16.
__device__ __forceinline__ v16h mk_a_f32(const float* row, int base,
                                         float scale) {
  const float4* p0 = (const float4*)(row + base);
  const float4* p1 = (const float4*)(row + base + 16);
  float4 a = p0[0], b = p0[1], c = p1[0], d = p1[1];
  v16h A;
  A[0] = (_Float16)(scale * a.x);  A[1] = (_Float16)(scale * a.y);
  A[2] = (_Float16)(scale * a.z);  A[3] = (_Float16)(scale * a.w);
  A[4] = (_Float16)(scale * b.x);  A[5] = (_Float16)(scale * b.y);
  A[6] = (_Float16)(scale * b.z);  A[7] = (_Float16)(scale * b.w);
  A[8] = (_Float16)(scale * c.x);  A[9] = (_Float16)(scale * c.y);
  A[10] = (_Float16)(scale * c.z); A[11] = (_Float16)(scale * c.w);
  A[12] = (_Float16)(scale * d.x); A[13] = (_Float16)(scale * d.y);
  A[14] = (_Float16)(scale * d.z); A[15] = (_Float16)(scale * d.w);
  return A;
}

// monotonic float<->uint mapping for atomicMax on floats
__device__ __forceinline__ unsigned fkey(float f) {
  int i = __float_as_int(f);
  return (i >= 0) ? (unsigned(i) | 0x80000000u) : ~unsigned(i);
}
__device__ __forceinline__ float funkey(unsigned u) {
  int i = (u & 0x80000000u) ? int(u & 0x7fffffffu) : ~int(u);
  return __int_as_float(i);
}

// ---------------------------------------------------------------------------
// Kernel 0: zero atomic accumulation regions of ws (poisoned by harness)
// ---------------------------------------------------------------------------
__global__ void init_kernel(double* sums, unsigned* hkey, float* ksumg) {
  int tid = threadIdx.x + blockIdx.x * blockDim.x;
  if (tid < 4) sums[tid] = 0.0;
  if (tid < HEADS) hkey[tid] = 0u;
  for (int i = tid; i < HEADS * DIM; i += blockDim.x * gridDim.x)
    ksumg[i] = 0.0f;
}

// ---------------------------------------------------------------------------
// Kernel 1: global sum / sumsq of q and k (f32 partials -> f64) +
//           per-head max of k
// ---------------------------------------------------------------------------
__global__ void __launch_bounds__(256)
stats_kernel(const float* __restrict__ q, const float* __restrict__ k,
             double* __restrict__ sums, unsigned* __restrict__ hkey) {
  const int head  = blockIdx.x / CHUNKS;
  const int chunk = blockIdx.x % CHUNKS;
  const size_t base = (size_t)head * HSTR + (size_t)chunk * (HSTR / CHUNKS);
  const float4* q4 = (const float4*)(q + base);
  const float4* k4 = (const float4*)(k + base);
  const int tid = threadIdx.x;

  double sq = 0.0, sq2 = 0.0, sk = 0.0, sk2 = 0.0;
  float km = -3.0e38f;
  for (int i = tid; i < (HSTR / CHUNKS) / 4; i += 256) {
    float4 a = q4[i];
    sq  += (double)((a.x + a.y) + (a.z + a.w));
    sq2 += (double)fmaf(a.x, a.x, fmaf(a.y, a.y, fmaf(a.z, a.z, a.w * a.w)));
    float4 b = k4[i];
    sk  += (double)((b.x + b.y) + (b.z + b.w));
    sk2 += (double)fmaf(b.x, b.x, fmaf(b.y, b.y, fmaf(b.z, b.z, b.w * b.w)));
    km = fmaxf(km, fmaxf(fmaxf(b.x, b.y), fmaxf(b.z, b.w)));
  }
  for (int off = 16; off >= 1; off >>= 1) {
    sq  += __shfl_xor(sq,  off, 32);
    sq2 += __shfl_xor(sq2, off, 32);
    sk  += __shfl_xor(sk,  off, 32);
    sk2 += __shfl_xor(sk2, off, 32);
    km = fmaxf(km, __shfl_xor(km, off, 32));
  }
  if ((tid & 31) == 0) {
    atomicAdd(&sums[0], sq);
    atomicAdd(&sums[1], sq2);
    atomicAdd(&sums[2], sk);
    atomicAdd(&sums[3], sk2);
    atomicMax(&hkey[head], fkey(km));
  }
}

// ---------------------------------------------------------------------------
// Kernel 2: alpha/beta scalars + decode per-head maxes
// ---------------------------------------------------------------------------
__global__ void finalize_kernel(const double* __restrict__ sums,
                                const unsigned* __restrict__ hkey,
                                float* __restrict__ scal,
                                float* __restrict__ hm) {
  int tid = threadIdx.x;
  if (tid == 0) {
    const double M = (double)BATCH * NH * SEQ * DIM;
    double varq = (sums[1] - sums[0] * sums[0] / M) / (M - 1.0);
    double vark = (sums[3] - sums[2] * sums[2] / M) / (M - 1.0);
    double st2  = (varq * vark - B_CONST) / (2.0 * A_CONST);
    double sigt = sqrt(st2);
    scal[0] = (float)(sigt / sqrt(varq));  // alpha
    scal[1] = (float)(sigt / sqrt(vark));  // beta
  }
  if (tid < HEADS) hm[tid] = funkey(hkey[tid]);
}

// ---------------------------------------------------------------------------
// Kernel 3: per head  KV[d][e] = sum_n exp(beta*k - bm)[n,d] * v[n,e]
//           and Ksum[d] = sum_n exp(beta*k - bm)[n,d]
// 1 workgroup / head, 16 waves; each wave owns one 16x16 KV tile (WMMA).
// Staged transposed in LDS so both WMMA operands read contiguous halves.
// ---------------------------------------------------------------------------
__global__ void __launch_bounds__(512)
kv_kernel(const float* __restrict__ k, const float* __restrict__ v,
          float* __restrict__ kvg, float* __restrict__ ksumg,
          const float* __restrict__ scal, const float* __restrict__ hm) {
  __shared__ _Float16 kesT[64][ROWP];  // kesT[d][n] = exp(beta*k[n,d]-bm)
  __shared__ _Float16 vshT[64][ROWP];  // vshT[e][n] = v[n,e]

  const int head = blockIdx.x;
  const int tid  = threadIdx.x;
  const int lane = tid & 31;
  const int wave = tid >> 5;                 // 0..15
  const int l15  = lane & 15;
  const int hi8  = (lane & 16) ? 8 : 0;
  const int hi16 = hi8 * 2;
  const int d0 = (wave >> 2) * 16;           // KV row-tile
  const int e0 = (wave & 3) * 16;            // KV col-tile
  const int cq = (tid & 15) * 4;             // owned column base (d/e index)
  const int r0 = tid >> 4;                   // 0..31 (row within chunk)

  const float beta = scal[1];
  const float bm   = beta * hm[head];
  const float* kh = k + (size_t)head * HSTR;
  const float* vh = v + (size_t)head * HSTR;

  float4 ksl = {0.f, 0.f, 0.f, 0.f};
  v8f acc = {0.f, 0.f, 0.f, 0.f, 0.f, 0.f, 0.f, 0.f};

  for (int nb = 0; nb < SEQ; nb += 64) {
    // prefetch next chunk while this one is processed
    if (nb + 64 < SEQ) {
      __builtin_prefetch(kh + (size_t)(nb + 64) * DIM + (size_t)tid * 8, 0, 1);
      __builtin_prefetch(vh + (size_t)(nb + 64) * DIM + (size_t)tid * 8, 0, 1);
    }
    __syncthreads();
#pragma unroll
    for (int rr = 0; rr < 2; ++rr) {
      int r = r0 + rr * 32;
      float4 kv4 = ((const float4*)(kh + (size_t)(nb + r) * DIM))[tid & 15];
      float4 e4;
      e4.x = __expf(beta * kv4.x - bm);
      e4.y = __expf(beta * kv4.y - bm);
      e4.z = __expf(beta * kv4.z - bm);
      e4.w = __expf(beta * kv4.w - bm);
      kesT[cq + 0][r] = (_Float16)e4.x;
      kesT[cq + 1][r] = (_Float16)e4.y;
      kesT[cq + 2][r] = (_Float16)e4.z;
      kesT[cq + 3][r] = (_Float16)e4.w;
      ksl.x += e4.x; ksl.y += e4.y; ksl.z += e4.z; ksl.w += e4.w;
      float4 vv4 = ((const float4*)(vh + (size_t)(nb + r) * DIM))[tid & 15];
      vshT[cq + 0][r] = (_Float16)vv4.x;
      vshT[cq + 1][r] = (_Float16)vv4.y;
      vshT[cq + 2][r] = (_Float16)vv4.z;
      vshT[cq + 3][r] = (_Float16)vv4.w;
    }
    __syncthreads();
#pragma unroll
    for (int nc = 0; nc < 64; nc += 32) {
      // A[d, nn] = kesT[d0+l15][nc + a_kk]: runs at nc+hi8 and nc+hi8+16
      v16h A = ld2x8(&kesT[d0 + l15][nc + hi8], &kesT[d0 + l15][nc + hi8 + 16]);
      // B[nn, e] = vshT[e0+l15][nc + b_kk]: runs at nc+hi8 and nc+hi8+16
      v16h Bm = ld2x8(&vshT[e0 + l15][nc + hi8], &vshT[e0 + l15][nc + hi8 + 16]);
      acc = wmma16x16x32(A, Bm, acc);
    }
  }
#pragma unroll
  for (int r = 0; r < 8; ++r)
    kvg[(size_t)head * (DIM * DIM) + (size_t)(d0 + r + hi8) * DIM + e0 + l15] =
        acc[r];
  atomicAdd(&ksumg[head * DIM + cq + 0], ksl.x);
  atomicAdd(&ksumg[head * DIM + cq + 1], ksl.y);
  atomicAdd(&ksumg[head * DIM + cq + 2], ksl.z);
  atomicAdd(&ksumg[head * DIM + cq + 3], ksl.w);
}

// ---------------------------------------------------------------------------
// Kernel 4: fused: block-diag SDPA + linear-attn output.
// Each workgroup (4 waves) handles BLK_PER_WG consecutive 64-token blocks of
// one head, staging KV/Ksum once per workgroup.
// ---------------------------------------------------------------------------
__global__ void __launch_bounds__(128)
fused_kernel(const float* __restrict__ q, const float* __restrict__ k,
             const float* __restrict__ v, const float* __restrict__ kvg,
             const float* __restrict__ ksumg, const float* __restrict__ scal,
             float* __restrict__ out) {
  __shared__ float    qs[64][64];         // 16 KB raw q (f32, feeds exp)
  __shared__ _Float16 ks[64][64];         // ks[j][d] (scores B reads rows)
  __shared__ _Float16 vsT[64][ROWP];      // vsT[e][j] = v[j,e]
  __shared__ _Float16 kvsT[64][ROWP];     // kvsT[e][d] = KV[d,e]
  __shared__ _Float16 ps[4][16][64];      // softmax P staging (per wave)
  __shared__ float    ksum_s[64];
  __shared__ float    ssum[4][16];        // per-row linear denominator

  const int head = blockIdx.x >> 3;            // 64 heads
  const int bq   = blockIdx.x & 7;             // block-group within head
  const int tid  = threadIdx.x;
  const int lane = tid & 31;
  const int wave = tid >> 5;
  const int l15  = lane & 15;
  const int hi8  = (lane & 16) ? 8 : 0;
  const int hi16 = hi8 * 2;
  const int m0   = wave * 16;

  const float alpha = scal[0];
  const float* kvh = kvg + (size_t)head * (DIM * DIM);

  // ---- stage KV (transposed) + Ksum once per workgroup ----
  for (int i = tid; i < (DIM * DIM) / 4; i += 128) {
    int r  = i >> 4;
    int c4 = (i & 15) * 4;
    float4 kvb = ((const float4*)kvh)[i];
    kvsT[c4 + 0][r] = (_Float16)kvb.x;
    kvsT[c4 + 1][r] = (_Float16)kvb.y;
    kvsT[c4 + 2][r] = (_Float16)kvb.z;
    kvsT[c4 + 3][r] = (_Float16)kvb.w;
  }
  if (tid < 64) ksum_s[tid] = ksumg[head * DIM + tid];
  __syncthreads();

  for (int bi = 0; bi < BLK_PER_WG; ++bi) {
    const int n0 = (bq * BLK_PER_WG + bi) * 64;
    const float* qh = q + (size_t)head * HSTR + (size_t)n0 * DIM;
    const float* kh = k + (size_t)head * HSTR + (size_t)n0 * DIM;
    const float* vh = v + (size_t)head * HSTR + (size_t)n0 * DIM;

    if (bi + 1 < BLK_PER_WG) {
      __builtin_prefetch(qh + DIM * 64 + (size_t)tid * 32, 0, 1);
      __builtin_prefetch(kh + DIM * 64 + (size_t)tid * 32, 0, 1);
      __builtin_prefetch(vh + DIM * 64 + (size_t)tid * 32, 0, 1);
    }

    // ---- stage q (f32), k (f16 rows), v (f16 transposed) ----
    for (int i = tid; i < (DIM * DIM) / 4; i += 128) {
      int r  = i >> 4;
      int c4 = (i & 15) * 4;
      *(float4*)&qs[r][c4] = ((const float4*)qh)[i];
      float4 kb = ((const float4*)kh)[i];
      v4h k4 = {(_Float16)kb.x, (_Float16)kb.y, (_Float16)kb.z, (_Float16)kb.w};
      *(v4h*)&ks[r][c4] = k4;
      float4 vb = ((const float4*)vh)[i];
      vsT[c4 + 0][r] = (_Float16)vb.x;
      vsT[c4 + 1][r] = (_Float16)vb.y;
      vsT[c4 + 2][r] = (_Float16)vb.z;
      vsT[c4 + 3][r] = (_Float16)vb.w;
    }
    __syncthreads();

    const float* qrow = &qs[m0 + l15][0];

    // ---- diag scores: S[i,j] = sum_d (q[i,d]/8) * k[j,d] ----
    v16h Aq0 = mk_a_f32(qrow, hi8, 0.125f);        // d-chunk 0
    v16h Aq1 = mk_a_f32(qrow, 32 + hi8, 0.125f);   // d-chunk 1
    v8f sc[4];
#pragma unroll
    for (int t = 0; t < 4; ++t) {
      v8f acc = {0.f, 0.f, 0.f, 0.f, 0.f, 0.f, 0.f, 0.f};
      acc = wmma16x16x32(Aq0, ld2x8(&ks[t * 16 + l15][hi8], &ks[t * 16 + l15][hi8 + 16]), acc);
      acc = wmma16x16x32(Aq1, ld2x8(&ks[t * 16 + l15][32 + hi8], &ks[t * 16 + l15][32 + hi8 + 16]), acc);
      sc[t] = acc;
    }

    // ---- row softmax (row of C tile = one half-wave in one VGPR) ----
    float rmax[8], rinv[8];
#pragma unroll
    for (int r = 0; r < 8; ++r) {
      float m = sc[0][r];
#pragma unroll
      for (int t = 1; t < 4; ++t) m = fmaxf(m, sc[t][r]);
      for (int off = 8; off >= 1; off >>= 1)
        m = fmaxf(m, __shfl_xor(m, off, 32));
      rmax[r] = m;
    }
#pragma unroll
    for (int r = 0; r < 8; ++r) {
      float s = 0.0f;
#pragma unroll
      for (int t = 0; t < 4; ++t) {
        float e = __expf(sc[t][r] - rmax[r]);
        sc[t][r] = e;
        s += e;
      }
      for (int off = 8; off >= 1; off >>= 1) s += __shfl_xor(s, off, 32);
      rinv[r] = 1.0f / s;
    }
#pragma unroll
    for (int t = 0; t < 4; ++t)
#pragma unroll
      for (int r = 0; r < 8; ++r)
        ps[wave][r + hi8][t * 16 + l15] = (_Float16)(sc[t][r] * rinv[r]);

    // ---- linear-attention Q operand + denominator (A-layout order) ----
    float xq[32], kss[32];
#pragma unroll
    for (int c = 0; c < 2; ++c)
#pragma unroll
      for (int rr = 0; rr < 2; ++rr) {
        int base = c * 32 + hi8 + rr * 16;
        const float4* p  = (const float4*)&qrow[base];
        const float4* p2 = (const float4*)&ksum_s[base];
        float4 a0 = p[0], a1 = p[1];
        float4 s0 = p2[0], s1 = p2[1];
        int o = c * 16 + rr * 8;
        xq[o + 0] = a0.x; xq[o + 1] = a0.y; xq[o + 2] = a0.z; xq[o + 3] = a0.w;
        xq[o + 4] = a1.x; xq[o + 5] = a1.y; xq[o + 6] = a1.z; xq[o + 7] = a1.w;
        kss[o + 0] = s0.x; kss[o + 1] = s0.y;
        kss[o + 2] = s0.z; kss[o + 3] = s0.w;
        kss[o + 4] = s1.x; kss[o + 5] = s1.y;
        kss[o + 6] = s1.z; kss[o + 7] = s1.w;
      }
    float rowm = -3.0e38f;
#pragma unroll
    for (int i = 0; i < 32; ++i) rowm = fmaxf(rowm, alpha * xq[i]);
    rowm = fmaxf(rowm, __shfl_xor(rowm, 16, 32));
    v16h aQ[2];
    float S = 0.0f;
#pragma unroll
    for (int i = 0; i < 32; ++i) {
      float e = __expf(alpha * xq[i] - rowm);
      aQ[i >> 4][i & 15] = (_Float16)e;
      S += e * kss[i];
    }
    S += __shfl_xor(S, 16, 32);
    if (lane < 16) ssum[wave][lane] = S + EPSF;
    __syncthreads();

    // ---- output tiles: 0.5 * (Qe*KV/S + P*V) ----
    const _Float16* prow = &ps[wave][l15][0];
    v16h Ap0 = ld2x8(&prow[hi8], &prow[hi8 + 16]);
    v16h Ap1 = ld2x8(&prow[32 + hi8], &prow[32 + hi8 + 16]);
    float* oh = out + (size_t)head * HSTR + (size_t)n0 * DIM;
#pragma unroll
    for (int et = 0; et < 4; ++et) {
      const _Float16* vrow  = &vsT[et * 16 + l15][0];
      const _Float16* kvrow = &kvsT[et * 16 + l15][0];
      v8f accD = {0.f, 0.f, 0.f, 0.f, 0.f, 0.f, 0.f, 0.f};
      v8f accL = {0.f, 0.f, 0.f, 0.f, 0.f, 0.f, 0.f, 0.f};
      accD = wmma16x16x32(Ap0, ld2x8(&vrow[hi8], &vrow[hi8 + 16]), accD);
      accD = wmma16x16x32(Ap1, ld2x8(&vrow[32 + hi8], &vrow[32 + hi8 + 16]), accD);
      accL = wmma16x16x32(aQ[0], ld2x8(&kvrow[hi8], &kvrow[hi8 + 16]), accL);
      accL = wmma16x16x32(aQ[1], ld2x8(&kvrow[32 + hi8], &kvrow[32 + hi8 + 16]), accL);
#pragma unroll
      for (int r = 0; r < 8; ++r) {
        float sden = ssum[wave][r + hi8];
        float o = 0.5f * (accL[r] / sden + accD[r]);
        oh[(size_t)(m0 + r + hi8) * DIM + et * 16 + l15] = o;
      }
    }
    __syncthreads();  // protect qs/ks/vsT before next block's staging
  }
}

// ---------------------------------------------------------------------------
extern "C" void kernel_launch(void* const* d_in, const int* in_sizes, int n_in,
                              void* d_out, int out_size, void* d_ws,
                              size_t ws_size, hipStream_t stream) {
  (void)in_sizes; (void)n_in; (void)out_size; (void)ws_size;
  const float* q = (const float*)d_in[0];
  const float* k = (const float*)d_in[1];
  const float* v = (const float*)d_in[2];
  float* out = (float*)d_out;

  char* w = (char*)d_ws;
  double*   sums  = (double*)w;                 // 4 doubles   [0,32)
  unsigned* hkey  = (unsigned*)(w + 32);        // 64 u32      [32,288)
  float*    scal  = (float*)(w + 288);          // alpha,beta  [288,304)
  float*    hm    = (float*)(w + 304);          // 64 floats   [304,560)
  float*    ksumg = (float*)(w + 1024);         // 64*64 f32   [1024,17408)
  float*    kvg   = (float*)(w + 1024 + 4096 * 4);  // 64*64*64 f32 (~1MB)

  init_kernel<<<16, 256, 0, stream>>>(sums, hkey, ksumg);
  stats_kernel<<<HEADS * CHUNKS, 256, 0, stream>>>(q, k, sums, hkey);
  finalize_kernel<<<1, 64, 0, stream>>>(sums, hkey, scal, hm);
  kv_kernel<<<HEADS, 512, 0, stream>>>(k, v, kvg, ksumg, scal, hm);
  fused_kernel<<<HEADS * BLK_PER_WG, 128, 0, stream>>>(q, k, v, kvg, ksumg,
                                                       scal, out);
}
